// Transformer_22668837388656
// MI455X (gfx1250) — compile-verified, matrix operand order checked by reading, not run on hardware
//
#include <hip/hip_runtime.h>

typedef unsigned short u16;
typedef unsigned int   u32;
typedef unsigned long long u64;
typedef __attribute__((ext_vector_type(8)))  float   v8f;
typedef __attribute__((ext_vector_type(16))) __bf16  v16bf;
typedef int v4i __attribute__((vector_size(16)));   // matches builtin param type

namespace cfg {
constexpr int B = 2, N = 1024, D = 1024, H = 16, DH = 64, L = 6;
constexpr int INNER = 2730, N2I = 5460;
constexpr int KKV = N + 1;      // 1025 keys incl. null slot
constexpr int KP  = 1056;       // KKV padded to multiple of 32
constexpr int IP  = 2752;       // INNER padded to multiple of 32
constexpr int TOK = B * N;      // 2048
}

// ---------- address-space casts (flat LDS addr low 32 bits == LDS byte address) ----------
typedef __attribute__((address_space(1))) v4i* gv4i_p;   // global (AS1) int4*
typedef __attribute__((address_space(3))) v4i* lv4i_p;   // LDS (AS3) int4*
__device__ __forceinline__ gv4i_p as1(const void* p) { return (gv4i_p)(u64)p; }
__device__ __forceinline__ lv4i_p as3(void* p)       { return (lv4i_p)(u32)(u64)p; }

#if defined(__has_builtin)
#if __has_builtin(__builtin_amdgcn_global_load_async_to_lds_b128)
#define HAVE_ASYNC_LDS 1
#endif
#endif

#ifdef HAVE_ASYNC_LDS
// per-lane 16B global -> LDS DMA, tracked by ASYNCcnt, no VGPR round trip
#define ASYNC_COPY16(gp, lp) __builtin_amdgcn_global_load_async_to_lds_b128(as1(gp), as3(lp), 0, 0)
#else
#define ASYNC_COPY16(gp, lp) (*(uint4*)(lp) = *(const uint4*)(gp))
#endif
#define WAIT_ASYNC() asm volatile("s_wait_asynccnt 0x0" ::: "memory")

// ---------- small helpers ----------
__device__ __forceinline__ u16 f2bf(float f) {
  u32 u = __float_as_uint(f);
  u32 r = (u + 0x7fffu + ((u >> 16) & 1u)) >> 16;   // RNE
  return (u16)r;
}
__device__ __forceinline__ float bf2f(u16 h) {
  return __uint_as_float(((u32)h) << 16);
}

__device__ __forceinline__ float block_reduce(float v, float* sm, bool do_max) {
  int tid = threadIdx.x;
#pragma unroll
  for (int o = 16; o > 0; o >>= 1) {               // wave32 butterfly
    float t = __shfl_xor(v, o, 32);
    v = do_max ? fmaxf(v, t) : (v + t);
  }
  if ((tid & 31) == 0) sm[tid >> 5] = v;
  __syncthreads();
  if (tid == 0) {
    float r = sm[0];
    int nw = blockDim.x >> 5;
    for (int i = 1; i < nw; ++i) r = do_max ? fmaxf(r, sm[i]) : (r + sm[i]);
    sm[0] = r;
  }
  __syncthreads();
  float r = sm[0];
  __syncthreads();
  return r;
}

// ---------- WMMA bf16 GEMM: C[M,N] = A[M,K] @ B[N,K]^T (+bias) ----------
// A row-major [M][lda], B row-major [Nm][ldb] (i.e. already transposed),
// double-buffered LDS staged with async global->LDS copies.
union Frag { uint4 q[2]; v16bf v; };

__global__ __launch_bounds__(256)
void k_gemm(const u16* __restrict__ A, const u16* __restrict__ Bm,
            float* __restrict__ C, const float* __restrict__ bias,
            int M, int Nm, int K, int lda, int ldb, int ldc,
            long sA, long sB, long sC)
{
  constexpr int TM = 128, TN = 128, TK = 32;
  __shared__ __align__(16) u16 As[2][TM][TK];
  __shared__ __align__(16) u16 Bs[2][TN][TK];

  const int z = blockIdx.z;
  A  += (long)z * sA;
  Bm += (long)z * sB;
  C  += (long)z * sC;

  const int tm = blockIdx.y * TM, tn = blockIdx.x * TN;
  const int tid  = threadIdx.x;
  const int wid  = tid >> 5, lane = tid & 31;
  const int wm   = (wid & 3) * 32;     // 4 waves along M
  const int wn   = (wid >> 2) * 64;    // 2 waves along N
  const int lr   = lane & 15;
  const int hi   = (lane >> 4) * 16;   // upper half-wave holds K 16..31

  // zero-prefill LDS rows that fall outside M/Nm (done once; K edges don't exist)
  {
    const uint4 zz = {0u, 0u, 0u, 0u};
#pragma unroll
    for (int it = 0; it < 2; ++it) {
      int idx = tid + it * 256, r = idx >> 2, c8 = (idx & 3) * 8;
      if (tm + r >= M)  { *(uint4*)&As[0][r][c8] = zz; *(uint4*)&As[1][r][c8] = zz; }
      if (tn + r >= Nm) { *(uint4*)&Bs[0][r][c8] = zz; *(uint4*)&Bs[1][r][c8] = zz; }
    }
  }

  auto stage = [&](int buf, int k0) {
#pragma unroll
    for (int it = 0; it < 2; ++it) {
      int idx = tid + it * 256;          // 0..511
      int r   = idx >> 2;                // tile row
      int c8  = (idx & 3) * 8;           // K sub-chunk (8 bf16 = 16B)
      if (tm + r < M)
        ASYNC_COPY16(A + (long)(tm + r) * lda + k0 + c8, &As[buf][r][c8]);
      if (tn + r < Nm)
        ASYNC_COPY16(Bm + (long)(tn + r) * ldb + k0 + c8, &Bs[buf][r][c8]);
    }
  };

  v8f acc[2][4] = {};

  stage(0, 0);
  WAIT_ASYNC();
  __syncthreads();

  const int nk = K / TK;
  for (int t = 0; t < nk; ++t) {
    const int buf = t & 1;
    if (t + 1 < nk) stage(buf ^ 1, (t + 1) * TK);   // overlap next-tile DMA with WMMAs

    Frag a[2], b[4];
#pragma unroll
    for (int fm = 0; fm < 2; ++fm) {
      const uint4* p = (const uint4*)&As[buf][wm + fm * 16 + lr][hi];
      a[fm].q[0] = p[0]; a[fm].q[1] = p[1];
    }
#pragma unroll
    for (int fn = 0; fn < 4; ++fn) {
      const uint4* p = (const uint4*)&Bs[buf][wn + fn * 16 + lr][hi];
      b[fn].q[0] = p[0]; b[fn].q[1] = p[1];
    }
#pragma unroll
    for (int fm = 0; fm < 2; ++fm)
#pragma unroll
      for (int fn = 0; fn < 4; ++fn)
        acc[fm][fn] = __builtin_amdgcn_wmma_f32_16x16x32_bf16(
            false, a[fm].v, false, b[fn].v, (short)0, acc[fm][fn], false, false);

    WAIT_ASYNC();          // next buffer's DMA complete (this wave)
    __syncthreads();       // all waves done reading buf & see new data
  }

  // ---- epilogue: 16x16 f32 C layout (lanes 0-15: M=r, lanes 16-31: M=r+8) ----
  const int rofs = (lane >> 4) * 8;
  const int cofs = lane & 15;
#pragma unroll
  for (int fm = 0; fm < 2; ++fm)
#pragma unroll
    for (int fn = 0; fn < 4; ++fn) {
      int col = tn + wn + fn * 16 + cofs;
      if (col >= Nm) continue;
      float bb = bias ? bias[col] : 0.0f;
#pragma unroll
      for (int r = 0; r < 8; ++r) {
        int row = tm + wm + fm * 16 + rofs + r;
        if (row < M) C[(long)row * ldc + col] = acc[fm][fn][r] + bb;
      }
    }
}

// ---------- elementwise / reduction kernels ----------
__global__ void k_copy_f32(const float* __restrict__ in, float* __restrict__ out, long n) {
  long i = (long)blockIdx.x * blockDim.x + threadIdx.x;
  if (i < n) out[i] = in[i];
}

// f32 [R_in][C] -> bf16 transposed [C][R_out], rows >= R_in zero-padded
__global__ void k_cvt_bf16_t(const float* __restrict__ in, u16* __restrict__ out,
                             int R_in, int R_out, int Ccols) {
  long i = (long)blockIdx.x * blockDim.x + threadIdx.x;
  long total = (long)R_out * Ccols;
  if (i >= total) return;
  int r = (int)(i % R_out);           // K index
  int c = (int)(i / R_out);           // N index
  out[i] = (r < R_in) ? f2bf(in[(long)r * Ccols + c]) : (u16)0;
}

__global__ __launch_bounds__(256)
void k_ln_bf16(const float* __restrict__ x, const float* __restrict__ g,
               const float* __restrict__ bta, u16* __restrict__ out) {
  __shared__ float sm[8];
  const int cols = cfg::D;
  long row = blockIdx.x;
  const float* xr = x + row * cols;
  float v[4], s = 0.f;
#pragma unroll
  for (int i = 0; i < 4; ++i) { v[i] = xr[threadIdx.x + i * 256]; s += v[i]; }
  float mean = block_reduce(s, sm, false) * (1.0f / cols);
  float q = 0.f;
#pragma unroll
  for (int i = 0; i < 4; ++i) { float d = v[i] - mean; q += d * d; }
  float rstd = rsqrtf(block_reduce(q, sm, false) * (1.0f / cols) + 1e-5f);
#pragma unroll
  for (int i = 0; i < 4; ++i) {
    int c = threadIdx.x + i * 256;
    out[row * cols + c] = f2bf((v[i] - mean) * rstd * g[c] + bta[c]);
  }
}

__global__ __launch_bounds__(256)
void k_add_ln(float* __restrict__ x, const float* __restrict__ t,
              const float* __restrict__ g, const float* __restrict__ bta) {
  __shared__ float sm[8];
  const int cols = cfg::D;
  long row = blockIdx.x;
  const float* tr = t + row * cols;
  float v[4], s = 0.f;
#pragma unroll
  for (int i = 0; i < 4; ++i) { v[i] = tr[threadIdx.x + i * 256]; s += v[i]; }
  float mean = block_reduce(s, sm, false) * (1.0f / cols);
  float q = 0.f;
#pragma unroll
  for (int i = 0; i < 4; ++i) { float d = v[i] - mean; q += d * d; }
  float rstd = rsqrtf(block_reduce(q, sm, false) * (1.0f / cols) + 1e-5f);
#pragma unroll
  for (int i = 0; i < 4; ++i) {
    int c = threadIdx.x + i * 256;
    x[row * cols + c] += (v[i] - mean) * rstd * g[c] + bta[c];
  }
}

__global__ __launch_bounds__(256)
void k_final_ln(const float* __restrict__ x, const float* __restrict__ g,
                const float* __restrict__ bta, float* __restrict__ out) {
  __shared__ float sm[8];
  const int cols = cfg::D;
  long row = blockIdx.x;
  const float* xr = x + row * cols;
  float v[4], mx = -3.402823466e38f;
#pragma unroll
  for (int i = 0; i < 4; ++i) { v[i] = xr[threadIdx.x + i * 256]; mx = fmaxf(mx, v[i]); }
  mx = block_reduce(mx, sm, true);
  float inv = 1.0f / mx, s = 0.f;
#pragma unroll
  for (int i = 0; i < 4; ++i) { v[i] *= inv; s += v[i]; }
  float mean = block_reduce(s, sm, false) * (1.0f / cols);
  float q = 0.f;
#pragma unroll
  for (int i = 0; i < 4; ++i) { float d = v[i] - mean; q += d * d; }
  float rstd = rsqrtf(block_reduce(q, sm, false) * (1.0f / cols) + 1e-5f);
#pragma unroll
  for (int i = 0; i < 4; ++i) {
    int c = threadIdx.x + i * 256;
    out[row * cols + c] = (v[i] - mean) * rstd * g[c] + bta[c];
  }
}

__global__ void k_pack_q(const float* __restrict__ q, u16* __restrict__ qb) {
  int idx = blockIdx.x * blockDim.x + threadIdx.x;
  const int total = cfg::TOK * cfg::D;
  if (idx >= total) return;
  int d = idx & 63, h = (idx >> 6) & 15, n = (idx >> 10) & 1023, b = idx >> 20;
  qb[(((long)(b * cfg::H + h) * cfg::N + n) * cfg::DH) + d] = f2bf(q[idx] * 0.125f); // DH^-0.5
}

// K rows stay [j][d]; V written transposed [d][j] so PV GEMM is also the BT form.
__global__ void k_pack_kv(const float* __restrict__ kv, const float* __restrict__ nk,
                          const float* __restrict__ nv,
                          u16* __restrict__ kb, u16* __restrict__ vbT) {
  int idx = blockIdx.x * blockDim.x + threadIdx.x;
  const int total = cfg::B * cfg::H * cfg::KP * cfg::DH;
  if (idx >= total) return;
  int d = idx & 63;
  int j = (idx >> 6) % cfg::KP;
  int h = (idx / (64 * cfg::KP)) % cfg::H;
  int b = idx / (64 * cfg::KP * cfg::H);
  float kvl = 0.f, vvl = 0.f;
  if (j == 0) { kvl = nk[h * 64 + d]; vvl = nv[h * 64 + d]; }
  else if (j <= cfg::N) {
    long base = ((long)b * cfg::N + (j - 1)) * (2 * cfg::D);
    kvl = kv[base + h * 64 + d];
    vvl = kv[base + cfg::D + h * 64 + d];
  }
  long zh = (long)(b * cfg::H + h);
  kb [(zh * cfg::KP + j) * cfg::DH + d] = f2bf(kvl);
  vbT[(zh * cfg::DH + d) * cfg::KP + j] = f2bf(vvl);
}

__global__ __launch_bounds__(256)
void k_softmax(const float* __restrict__ S, u16* __restrict__ P) {
  __shared__ float sm[8];
  int i = blockIdx.x;           // query index
  int z = blockIdx.y;           // b*H + h
  const float* sr = S + ((long)z * cfg::N + i) * cfg::KP;
  u16*         pr = P + ((long)z * cfg::N + i) * cfg::KP;
  int lim = i + 2;              // causal: j <= i+1 allowed (null slot j=0 always)
  float mx = -3.402823466e38f;
  for (int j = threadIdx.x; j < lim; j += 256) mx = fmaxf(mx, sr[j]);
  mx = block_reduce(mx, sm, true);
  float s = 0.f;
  for (int j = threadIdx.x; j < lim; j += 256) s += expf(sr[j] - mx);
  s = block_reduce(s, sm, false);
  float inv = 1.0f / s;
  for (int j = threadIdx.x; j < cfg::KP; j += 256)
    pr[j] = (j < lim) ? f2bf(expf(sr[j] - mx) * inv) : (u16)0;
}

__global__ __launch_bounds__(256)
void k_talk_mix(const u16* __restrict__ P, const float* __restrict__ talk,
                u16* __restrict__ PM) {
  __shared__ float tk[16][16];
  tk[threadIdx.x >> 4][threadIdx.x & 15] = talk[threadIdx.x];
  __syncthreads();
  long idx = (long)blockIdx.x * 256 + threadIdx.x;
  const long total = (long)cfg::B * cfg::N * cfg::KP;
  if (idx >= total) return;
  int j = (int)(idx % cfg::KP);
  int i = (int)((idx / cfg::KP) % cfg::N);
  int b = (int)(idx / ((long)cfg::KP * cfg::N));
  float p[16];
#pragma unroll
  for (int h = 0; h < 16; ++h)
    p[h] = bf2f(P[(((long)(b * 16 + h) * cfg::N + i) * cfg::KP) + j]);
#pragma unroll
  for (int g = 0; g < 16; ++g) {
    float acc = 0.f;
#pragma unroll
    for (int h = 0; h < 16; ++h) acc += tk[g][h] * p[h];
    PM[(((long)(b * 16 + g) * cfg::N + i) * cfg::KP) + j] = f2bf(acc);
  }
}

__global__ void k_permute_o(const float* __restrict__ o, u16* __restrict__ ob) {
  int idx = blockIdx.x * blockDim.x + threadIdx.x;
  const int total = cfg::TOK * cfg::D;
  if (idx >= total) return;
  int d = idx & 63, n = (idx >> 6) & 1023, h = (idx >> 16) & 15, b = idx >> 20;
  ob[((long)(b * cfg::N + n)) * cfg::D + h * 64 + d] = f2bf(o[idx]);
}

__global__ void k_geglu(const float* __restrict__ u, u16* __restrict__ gbuf) {
  long idx = (long)blockIdx.x * 256 + threadIdx.x;
  const long total = (long)cfg::TOK * cfg::IP;
  if (idx >= total) return;
  int  c = (int)(idx % cfg::IP);
  long r = idx / cfg::IP;
  u16 outv = 0;
  if (c < cfg::INNER) {
    float ua = u[r * cfg::N2I + c];
    float ug = u[r * cfg::N2I + cfg::INNER + c];
    float gl = ug * 0.5f * (1.0f + erff(ug * 0.70710678118654752f)); // exact GELU
    outv = f2bf(ua * gl);
  }
  gbuf[idx] = outv;
}

// ---------- host orchestration ----------
static void gemm(const u16* A, const u16* Bm, float* C, const float* bias,
                 int M, int Nm, int K, int lda, int ldb, int ldc,
                 long sA, long sB, long sC, int batch, hipStream_t s) {
  dim3 grid((Nm + 127) / 128, (M + 127) / 128, batch), blk(256);
  k_gemm<<<grid, blk, 0, s>>>(A, Bm, C, bias, M, Nm, K, lda, ldb, ldc, sA, sB, sC);
}

extern "C" void kernel_launch(void* const* d_in, const int* in_sizes, int n_in,
                              void* d_out, int out_size, void* d_ws, size_t ws_size,
                              hipStream_t stream) {
  using namespace cfg;
  const float* x_in    = (const float*)d_in[0];
  const float* apre_g  = (const float*)d_in[1];
  const float* apre_b  = (const float*)d_in[2];
  const float* apost_g = (const float*)d_in[3];
  const float* apost_b = (const float*)d_in[4];
  const float* Wq      = (const float*)d_in[5];
  const float* Wkv     = (const float*)d_in[6];
  const float* Wo      = (const float*)d_in[7];
  const float* bo      = (const float*)d_in[8];
  const float* null_k  = (const float*)d_in[9];
  const float* null_v  = (const float*)d_in[10];
  const float* talk    = (const float*)d_in[11];
  const float* fpre_g  = (const float*)d_in[12];
  const float* fpre_b  = (const float*)d_in[13];
  const float* fpost_g = (const float*)d_in[14];
  const float* fpost_b = (const float*)d_in[15];
  const float* W1      = (const float*)d_in[16];
  const float* b1      = (const float*)d_in[17];
  const float* W2      = (const float*)d_in[18];
  const float* b2      = (const float*)d_in[19];
  const float* fin_g   = (const float*)d_in[20];
  const float* fin_b   = (const float*)d_in[21];
  float* out = (float*)d_out;
  (void)in_sizes; (void)n_in; (void)out_size; (void)ws_size;

  char* ws = (char*)d_ws;
  size_t off = 0;
  auto alloc = [&](size_t bytes) -> void* {
    void* p = ws + off;
    off = (off + bytes + 255) & ~(size_t)255;
    return p;
  };
  float* xb   = (float*)alloc((size_t)TOK * D * 4);       // residual stream (f32)
  u16*   hb   = (u16*)  alloc((size_t)TOK * D * 2);       // LN output / permuted o (bf16)
  float* S1   = (float*)alloc((size_t)TOK * N2I * 4);     // q proj / ff1 output (f32)
  float* S2   = (float*)alloc((size_t)TOK * 2 * D * 4);   // kv / o / layer outputs (f32)
  u16*   qb   = (u16*)  alloc((size_t)B * H * N * DH * 2);
  u16*   kb   = (u16*)  alloc((size_t)B * H * KP * DH * 2);
  u16*   vbT  = (u16*)  alloc((size_t)B * H * DH * KP * 2); // V transposed [d][j]
  float* SC   = (float*)alloc((size_t)B * H * N * KP * 4);  // attention scores
  u16*   P    = (u16*)  alloc((size_t)B * H * N * KP * 2);  // softmax probs (bf16)
  u16*   PM   = (u16*)  alloc((size_t)B * H * N * KP * 2);  // talked-heads probs
  u16*   wqT  = (u16*)  alloc((size_t)D * D * 2);           // [N][K] bf16 weights
  u16*   wkvT = (u16*)  alloc((size_t)2 * D * D * 2);
  u16*   woT  = (u16*)  alloc((size_t)D * D * 2);
  u16*   w1T  = (u16*)  alloc((size_t)N2I * D * 2);
  u16*   w2T  = (u16*)  alloc((size_t)D * IP * 2);          // K padded to 2752
  u16*   gb   = (u16*)  alloc((size_t)TOK * IP * 2);        // GEGLU output (bf16, K-padded)

  auto blocks = [](long n, int bs) { return (unsigned)((n + bs - 1) / bs); };

  k_copy_f32<<<blocks((long)TOK * D, 256), 256, 0, stream>>>(x_in, xb, (long)TOK * D);

  for (int l = 0; l < L; ++l) {
    // --- per-layer weight conversion f32 -> bf16, transposed to [N][K] ---
    k_cvt_bf16_t<<<blocks((long)D * D, 256), 256, 0, stream>>>(Wq + (long)l * D * D, wqT, D, D, D);
    k_cvt_bf16_t<<<blocks((long)2 * D * D, 256), 256, 0, stream>>>(Wkv + (long)l * D * 2 * D, wkvT, D, D, 2 * D);
    k_cvt_bf16_t<<<blocks((long)D * D, 256), 256, 0, stream>>>(Wo + (long)l * D * D, woT, D, D, D);
    k_cvt_bf16_t<<<blocks((long)N2I * D, 256), 256, 0, stream>>>(W1 + (long)l * D * N2I, w1T, D, D, N2I);
    k_cvt_bf16_t<<<blocks((long)D * IP, 256), 256, 0, stream>>>(W2 + (long)l * INNER * D, w2T, INNER, IP, D);

    // --- attention ---
    k_ln_bf16<<<TOK, 256, 0, stream>>>(xb, apre_g + l * D, apre_b + l * D, hb);
    gemm(hb, wqT,  S1, nullptr, TOK, D,     D, D, D, D,     0, 0, 0, 1, stream);
    gemm(hb, wkvT, S2, nullptr, TOK, 2 * D, D, D, D, 2 * D, 0, 0, 0, 1, stream);
    k_pack_q <<<blocks((long)TOK * D, 256), 256, 0, stream>>>(S1, qb);
    k_pack_kv<<<blocks((long)B * H * KP * DH, 256), 256, 0, stream>>>(
        S2, null_k + l * H * DH, null_v + l * H * DH, kb, vbT);
    // scores = (Q*scale) @ K^T   (batched over B*H heads)
    gemm(qb, kb, SC, nullptr, N, KKV, DH, DH, DH, KP,
         (long)N * DH, (long)KP * DH, (long)N * KP, B * H, stream);
    dim3 gs(N, B * H);
    k_softmax <<<gs, 256, 0, stream>>>(SC, P);
    k_talk_mix<<<blocks((long)B * N * KP, 256), 256, 0, stream>>>(P, talk + l * H * H, PM);
    // O = PM @ V   (B = V^T [DH][KP], K padded to 1056 with zeros)
    gemm(PM, vbT, S2, nullptr, N, DH, KP, KP, KP, DH,
         (long)N * KP, (long)DH * KP, (long)N * DH, B * H, stream);
    k_permute_o<<<blocks((long)TOK * D, 256), 256, 0, stream>>>(S2, hb);
    gemm(hb, woT, S2, bo + l * D, TOK, D, D, D, D, D, 0, 0, 0, 1, stream);
    k_add_ln<<<TOK, 256, 0, stream>>>(xb, S2, apost_g + l * D, apost_b + l * D);

    // --- GEGLU feedforward ---
    k_ln_bf16<<<TOK, 256, 0, stream>>>(xb, fpre_g + l * D, fpre_b + l * D, hb);
    gemm(hb, w1T, S1, b1 + (long)l * N2I, TOK, N2I, D, D, D, N2I, 0, 0, 0, 1, stream);
    k_geglu<<<blocks((long)TOK * IP, 256), 256, 0, stream>>>(S1, gb);
    gemm(gb, w2T, S2, b2 + l * D, TOK, D, IP, IP, IP, D, 0, 0, 0, 1, stream);
    k_add_ln<<<TOK, 256, 0, stream>>>(xb, S2, fpost_g + l * D, fpost_b + l * D);
  }

  k_final_ln<<<TOK, 256, 0, stream>>>(xb, fin_g, fin_b, out);
}
